// Node2Edge_73486890434885
// MI455X (gfx1250) — matrix operand order repaired and checked by reading, not compiled, not run on hardware
//
#include <hip/hip_runtime.h>
#include <hip/hip_bf16.h>

// Shapes from reference: S=128, L=512, IN_DIM=256, PROJ=32, OUT=128
#define S_DIM 128
#define L_DIM 512
#define IN_D  256
#define PJ    32
#define OUTF  128

typedef __attribute__((ext_vector_type(16))) __bf16 v16bf;
typedef __attribute__((ext_vector_type(8)))  float  v8f;

union FragU { v16bf v; uint4 q[2]; };

__device__ __forceinline__ __bf16 f2bf(float f) { return (__bf16)f; }

// CDNA5 async global->LDS copy (16B per lane), tracked by ASYNCcnt.
__device__ __forceinline__ void async_copy16(unsigned lds_addr, const void* gptr) {
  asm volatile("global_load_async_to_lds_b128 %0, %1, off"
               :: "v"(lds_addr), "v"((unsigned long long)(uintptr_t)gptr)
               : "memory");
}
__device__ __forceinline__ void wait_async0() {
  asm volatile("s_wait_asynccnt 0" ::: "memory");
}
__device__ __forceinline__ unsigned lds_abs(const void* p) {
  return (unsigned)(uintptr_t)p;   // flat->LDS: addr[31:0] is the LDS byte address
}

// G panel LDS stride in bf16 elements (multiple of 8 -> 16B-aligned rows)
#define GSTRIDE 264

// ---------------------------------------------------------------------------
// Prep 1: LayerNorm + projection.  Emits:
//   Lt   : bf16 [L*32][S]                (A-matrix layout for stage A: rows=(i,d), K=s)
//   Rpack: bf16 [1024 ct][4 ks][32 lane][16 h]   (pre-packed stage-A B fragments)
// Block = 256 threads = 4 row-slots x 64 lanes; 16 rows per block.
// ---------------------------------------------------------------------------
__global__ __launch_bounds__(256, 1) void node2edge_prep(
    const float* __restrict__ x,       // [S][L][256]
    const float* __restrict__ mask,    // [S][L]
    const float* __restrict__ w_proj,  // [64][256]
    const float* __restrict__ b_proj,  // [64]
    __bf16* __restrict__ Lt,
    __bf16* __restrict__ Rpack)
{
  extern __shared__ char smem[];
  float* wlds  = (float*)smem;            // 64 x 257 (padded vs bank conflicts)
  float* xs    = wlds + 64 * 257;         // 4 x 256
  float* rsum  = xs + 4 * 256;            // 4 x 64
  float* rsq   = rsum + 4 * 64;           // 4 x 64
  float* stats = rsq + 4 * 64;            // 4 x 2  (mu, rsig)

  const int tid = threadIdx.x;

  for (int idx = tid; idx < 64 * 256; idx += 256) {
    int o = idx >> 8, c = idx & 255;
    wlds[o * 257 + c] = w_proj[idx];
  }
  __syncthreads();

  const int q = tid >> 6;   // row slot 0..3
  const int o = tid & 63;   // output channel / loader lane

  for (int it = 0; it < 4; ++it) {
    const int rid = blockIdx.x * 16 + it * 4 + q;   // flat (s,l) row
    const int s = rid >> 9;
    const int l = rid & 511;
    const float* xr = x + (long)rid * IN_D;

    float s1 = 0.f, s2 = 0.f;
#pragma unroll
    for (int k = 0; k < 4; ++k) {
      float v = xr[o * 4 + k];
      xs[q * 256 + o * 4 + k] = v;
      s1 += v; s2 += v * v;
    }
    rsum[q * 64 + o] = s1;
    rsq [q * 64 + o] = s2;
    __syncthreads();

    if (o == 0) {
      float a = 0.f, b = 0.f;
      for (int k = 0; k < 64; ++k) { a += rsum[q * 64 + k]; b += rsq[q * 64 + k]; }
      float mu  = a * (1.f / 256.f);
      float var = b * (1.f / 256.f) - mu * mu;
      stats[q * 2 + 0] = mu;
      stats[q * 2 + 1] = rsqrtf(var + 1e-5f);
    }
    __syncthreads();

    const float mu = stats[q * 2 + 0];
    const float rs = stats[q * 2 + 1];
    const float m  = mask[s * L_DIM + l];

    float dot = 0.f;
    for (int c = 0; c < 256; ++c)
      dot += (xs[q * 256 + c] - mu) * wlds[o * 257 + c];
    const float act = (dot * rs + b_proj[o]) * m;

    if (o < PJ) {
      Lt[(l * PJ + o) * S_DIM + s] = f2bf(act);
    } else {
      // Pack into stage-A B-fragment layout: k = s, n_global = l*32 + e.
      const int e  = o - PJ;
      const int ng = l * PJ + e;
      const int ct = ng >> 4;                          // 16-wide col tile
      const int ln = ((s >> 3) & 1) * 16 + (ng & 15); // lane = 16*((K>>3)&1) + N
      const int ks = s >> 5;
      const int h  = (s & 7) + ((s >> 4) & 1) * 8;     // half index = (K%8) + 8*((K>>4)&1)
      Rpack[(((long)ct * 4 + ks) * 32 + ln) * 16 + h] = f2bf(act);
    }
    __syncthreads();
  }
}

// ---------------------------------------------------------------------------
// Prep 2: pack out_weights (d,e,f) into stage-B B-fragments:
//   Wpack: bf16 [32 ks2][8 nt2][32 lane][16 h],  k2 = d*32+e, f = n.
// ---------------------------------------------------------------------------
__global__ __launch_bounds__(256, 1) void node2edge_packw(
    const float* __restrict__ w3,      // [32][32][128] = [k2][f]
    __bf16* __restrict__ Wpack)
{
  const int idx  = blockIdx.x * 256 + threadIdx.x;   // < 131072
  const int frag = idx >> 9;                         // ks2*8 + nt2
  const int elem = idx & 511;
  const int lane = elem >> 4;
  const int h    = elem & 15;
  const int ks2  = frag >> 3;
  const int nt2  = frag & 7;
  const int k2   = ks2 * 32 + (lane >> 4) * 8 + (h & 7) + (h >> 3) * 16;
  const int f    = nt2 * 16 + (lane & 15);
  Wpack[idx] = f2bf(w3[k2 * OUTF + f]);
}

// ---------------------------------------------------------------------------
// Main: one workgroup per 8x8 (i,j) output tile.
//   Stage A: G[(i,d),(j,e)] = sum_s Lt * r   (256x256, K=128) -> bf16 LDS
//   Stage B: out[(i,j),f]   = sum_{d,e} G * W (64x128, K=1024)
// B-fragments are staged into LDS once per workgroup with async copies.
// ---------------------------------------------------------------------------
__global__ __launch_bounds__(256, 1) void node2edge_main(
    const __bf16* __restrict__ Lt,       // [L*32][S]
    const __bf16* __restrict__ Rpack,    // packed stage-A B frags
    const __bf16* __restrict__ Wpack,    // packed stage-B B frags (256 KB)
    const float*  __restrict__ mask,     // [S][L]
    const float*  __restrict__ out_bias, // [128]
    float* __restrict__ out)             // [L][L][128]
{
  extern __shared__ char smem[];
  __bf16* Gs    = (__bf16*)smem;                             // 256 x GSTRIDE
  __bf16* Rs    = (__bf16*)(smem + 256 * GSTRIDE * 2);       // 32768 halves (64 KB)
  __bf16* Ws1   = (__bf16*)(smem + 256 * GSTRIDE * 2 + 65536); // 64 KB
  float*  norms = (float*)(smem + 256 * GSTRIDE * 2 + 2 * 65536); // 64

  const int tid   = threadIdx.x;
  const int lane  = tid & 31;
  const int wave  = tid >> 5;
  const int lhalf = lane & 15;
  const int lgrp  = lane >> 4;
  const int i0    = blockIdx.x * 8;
  const int j0    = blockIdx.y * 8;

  // ---- async-stage the 64KB Rpack slice for this j-tile into LDS ----
  {
    const char* src = (const char*)(Rpack + (size_t)blockIdx.y * 32768);
    const unsigned dst = lds_abs(Rs);
#pragma unroll
    for (int k = 0; k < 16; ++k) {
      const unsigned off = (unsigned)(tid + k * 256) * 16u;
      async_copy16(dst + off, src + off);
    }
  }

  // norm[i,j] = sum_s mask[s,i]*mask[s,j]  (overlaps the async copy)
  if (tid < 64) {
    const int il = tid >> 3, jl = tid & 7;
    const float* mi = mask + (i0 + il);
    const float* mj = mask + (j0 + jl);
    float acc = 0.f;
    for (int s = 0; s < S_DIM; ++s)
      acc += mi[s * L_DIM] * mj[s * L_DIM];
    norms[tid] = acc;
  }
  wait_async0();
  __syncthreads();

  // ---------------- Stage A: build G panel in LDS (bf16) ----------------
  for (int mtw = 0; mtw < 2; ++mtw) {
    const int mt   = wave * 2 + mtw;                  // row-tile 0..15
    const int rowA = i0 * PJ + mt * 16 + lhalf;       // global (i,d) row
    const __bf16* lrow = Lt + (long)rowA * S_DIM;

    FragU afr[4];
#pragma unroll
    for (int ks = 0; ks < 4; ++ks) {
      const uint4* p = (const uint4*)(lrow + ks * 32 + lgrp * 8);
      afr[ks].q[0] = p[0];        // K = g*8 .. g*8+7
      afr[ks].q[1] = p[2];        // K = 16+g*8 .. 16+g*8+7
    }

    for (int nt = 0; nt < 16; ++nt) {
      v8f acc = {};
#pragma unroll
      for (int ks = 0; ks < 4; ++ks) {
        FragU bfr;
        const uint4* bp = (const uint4*)(Rs + ((nt * 4 + ks) * 32 + lane) * 16);
        bfr.q[0] = bp[0];
        bfr.q[1] = bp[1];
        acc = __builtin_amdgcn_wmma_f32_16x16x32_bf16(
                  false, afr[ks].v, false, bfr.v, (short)0, acc, false, false);
      }
      const int col   = nt * 16 + lhalf;
      const int rbase = mt * 16 + lgrp * 8;
#pragma unroll
      for (int vr = 0; vr < 8; ++vr)
        Gs[(rbase + vr) * GSTRIDE + col] = f2bf(acc[vr]);
    }
  }
  __syncthreads();   // Gs visible; Rs region now dead -> reuse as W buffer 0

  // ---------------- Stage B: out = G x W  (W streamed via LDS dbl-buffer) ---
  const int mt2    = wave >> 1;          // row-tile over 64 (i,j) rows
  const int ntbase = (wave & 1) * 4;     // 4 f-tiles per wave
  v8f acc2[4] = {};

  const int m2 = mt2 * 16 + lhalf;       // (i_loc, j_loc) row
  const int il = m2 >> 3;
  const int jl = m2 & 7;
  const __bf16* grow = Gs + (il * PJ) * GSTRIDE + jl * PJ + lgrp * 8;

  __bf16* wbuf[2] = { Rs, Ws1 };

  // prologue: stage phase 0 (ks2 = 0..7)
  {
    const unsigned dst = lds_abs(wbuf[0]);
#pragma unroll
    for (int k = 0; k < 16; ++k) {
      const unsigned off = (unsigned)(tid + k * 256) * 16u;
      async_copy16(dst + off, (const char*)Wpack + off);
    }
    wait_async0();
  }
  __syncthreads();

  for (int p = 0; p < 4; ++p) {          // 4 phases x 8 ks2 steps
    if (p < 3) {                         // issue next phase (overlaps compute)
      const unsigned dst = lds_abs(wbuf[(p + 1) & 1]);
      const char* src = (const char*)Wpack + (size_t)(p + 1) * 65536;
#pragma unroll
      for (int k = 0; k < 16; ++k) {
        const unsigned off = (unsigned)(tid + k * 256) * 16u;
        async_copy16(dst + off, src + off);
      }
    }

    const __bf16* wsrc = wbuf[p & 1];
    for (int kk = 0; kk < 8; ++kk) {
      const int ks2 = p * 8 + kk;        // k-step == d index
      FragU a2;
      const __bf16* gp = grow + ks2 * GSTRIDE;
      a2.q[0] = *(const uint4*)(gp);       // e = g*8 .. g*8+7
      a2.q[1] = *(const uint4*)(gp + 16);  // e = 16+g*8 ..
#pragma unroll
      for (int u = 0; u < 4; ++u) {
        const int nt2 = ntbase + u;
        FragU b2;
        const uint4* wp = (const uint4*)(wsrc + ((kk * 8 + nt2) * 32 + lane) * 16);
        b2.q[0] = wp[0];
        b2.q[1] = wp[1];
        acc2[u] = __builtin_amdgcn_wmma_f32_16x16x32_bf16(
                      false, a2.v, false, b2.v, (short)0, acc2[u], false, false);
      }
    }

    if (p < 3) {
      wait_async0();      // next phase landed (had a full compute phase to fly)
      __syncthreads();    // all waves done reading current buffer
    }
  }

  // ---------------- Epilogue: bias, normalize, store ----------------
#pragma unroll
  for (int u = 0; u < 4; ++u) {
    const int f    = (ntbase + u) * 16 + lhalf;
    const float bb = out_bias[f];
#pragma unroll
    for (int vr = 0; vr < 8; ++vr) {
      const int m  = mt2 * 16 + lgrp * 8 + vr;
      const int iL = m >> 3;
      const int jL = m & 7;
      const float nv  = norms[iL * 8 + jL] + 0.001f;
      const float val = (acc2[u][vr] + bb) / nv;
      out[((long)(i0 + iL) * L_DIM + (j0 + jL)) * OUTF + f] = val;
    }
  }
}

// ---------------------------------------------------------------------------
extern "C" void kernel_launch(void* const* d_in, const int* in_sizes, int n_in,
                              void* d_out, int out_size, void* d_ws, size_t ws_size,
                              hipStream_t stream) {
  const float* node_repr = (const float*)d_in[0];  // 128*512*256
  const float* mask      = (const float*)d_in[1];  // 128*512
  const float* w_proj    = (const float*)d_in[2];  // 64*256
  const float* b_proj    = (const float*)d_in[3];  // 64
  const float* out_w     = (const float*)d_in[4];  // 32*32*128
  const float* out_bias  = (const float*)d_in[5];  // 128
  float* out = (float*)d_out;                      // 512*512*128

  char* ws = (char*)d_ws;
  __bf16* Lt    = (__bf16*)(ws);                   // 4 MB  : [16384][128]
  __bf16* Rpack = (__bf16*)(ws + (4u << 20));      // 4 MB  : packed B frags
  __bf16* Wpack = (__bf16*)(ws + (8u << 20));      // 256 KB: packed W frags

  const size_t lds1 = (size_t)(64 * 257 + 4 * 256 + 4 * 64 + 4 * 64 + 8) * sizeof(float);
  node2edge_prep<<<4096, 256, lds1, stream>>>(node_repr, mask, w_proj, b_proj, Lt, Rpack);

  node2edge_packw<<<512, 256, 0, stream>>>(out_w, Wpack);

  const size_t lds3 = (size_t)256 * GSTRIDE * sizeof(__bf16)  // Gs   (132 KB)
                    + 2 * 65536                                // Rs/Ws0 + Ws1
                    + 64 * sizeof(float);                      // norms
  node2edge_main<<<dim3(64, 64), 256, lds3, stream>>>(Lt, Rpack, Wpack, mask, out_bias, out);
}
